// MultiHeadAttention_44590350467332
// MI455X (gfx1250) — matrix operand order repaired and checked by reading, not compiled, not run on hardware
//
#include <hip/hip_runtime.h>
#include <hip/hip_bf16.h>
#include <math.h>

// ---------------------------------------------------------------------------
// MHA forward for MI455X (gfx1250): f16 WMMA (v_wmma_f32_16x16x32_f16) GEMMs,
// fp32 softmax, attn written to HBM exactly once with NT stores, PV GEMM fed
// from LDS. Projection GEMM register-double-buffers its LDS staging; B-tile
// transpose is staged k-contiguously so LDS stores are 2x b128 per thread.
// ---------------------------------------------------------------------------

typedef _Float16 h16;
typedef __attribute__((ext_vector_type(16))) _Float16 v16h;
typedef __attribute__((ext_vector_type(8)))  _Float16 h8;
typedef __attribute__((ext_vector_type(4)))  _Float16 h4;
typedef __attribute__((ext_vector_type(8)))  float    v8f;
typedef __attribute__((ext_vector_type(4)))  float    f4v;

#define D_MODEL   1024
#define NUM_HEADS 16
#define DK        64
#define SEQ       2048
#define BATCH     2
#define MTOT      (BATCH * SEQ)   // 4096 rows of the flattened [B*S, D] matrices

#define PS_LD 2052   // f32 score slab row stride (multiple of 4, pads banks)
#define PH_LD 2056   // f16 prob  slab row stride (4112 B rows, 16B aligned)

// --- WMMA wrapper -----------------------------------------------------------
__device__ __forceinline__ v8f wmma_f16(v16h a, v16h b, v8f c) {
  // 8 args: (neg_a, A, neg_b, B, c_mod, C, reuse_a, reuse_b)
  return __builtin_amdgcn_wmma_f32_16x16x32_f16(false, a, false, b, (short)0, c,
                                                false, false);
}

// --- A-matrix fragment: 16x32 f16 tile, rows contiguous with stride ldh -----
// ISA layout: lane row = lane&15; halves 0..7 -> K = (hi?8:0)+0..7,
//             halves 8..15 -> K = 16+(hi?8:0)+0..7    (hi = lane>=16)
__device__ __forceinline__ v16h load_frag_a(const h16* tile, int ldh) {
  int lane = threadIdx.x & 31;
  int row  = lane & 15;
  int hi   = lane >> 4;
  const h16* p = tile + (size_t)row * ldh + (hi ? 8 : 0);
  h8 lo = *(const h8*)(p);
  h8 up = *(const h8*)(p + 16);
  v16h a;
#pragma unroll
  for (int i = 0; i < 8; ++i) { a[i] = lo[i]; a[i + 8] = up[i]; }
  return a;
}

// --- B-matrix fragment: 32x16 f16 tile given column-major storage -----------
// colsBase + n*ldh points at the contiguous K-vector of column n.
// ISA layout: lane col = lane&15; halves 0..7 -> K = (hi?8:0)+0..7,
//             halves 8..15 -> K = 16+(hi?8:0)+0..7
__device__ __forceinline__ v16h load_frag_b(const h16* colsBase, int ldh,
                                            int kbase) {
  int lane = threadIdx.x & 31;
  int n    = lane & 15;
  int hi   = lane >> 4;
  const h16* p = colsBase + (size_t)n * ldh + kbase + (hi ? 8 : 0);
  h8 x0 = *(const h8*)(p);
  h8 x1 = *(const h8*)(p + 16);
  v16h b;
#pragma unroll
  for (int i = 0; i < 8; ++i) { b[i] = x0[i]; b[i + 8] = x1[i]; }
  return b;
}

// ---------------------------------------------------------------------------
// Projection GEMM: C[m][n] = sum_k A[m][k] * W[k][n] + bias[n]
//   mode 0: f32 row-major to outF (NT)   (final O projection)
//   mode 1: f16 head-major  [bh][s][d]   (Q, K)
//   mode 2: f16 transposed  [bh][d][s]   (V -> Vt, for PV B-fragments)
// Block: 64x64 tile, 128 threads (4 waves), each wave a 32x32 quadrant.
// ---------------------------------------------------------------------------
__global__ __launch_bounds__(128) void gemm_proj(
    const float* __restrict__ A, const float* __restrict__ W,
    const float* __restrict__ bias, float* __restrict__ outF,
    h16* __restrict__ outH, int M, int N, int K, int mode) {
  __shared__ h16 As[64 * 48];   // [row][k], stride 48 halfs (96B, 16B aligned)
  __shared__ h16 Bs[64 * 48];   // [n][k] (transposed), stride 48

  const int tid    = threadIdx.x;
  const int m_base = blockIdx.x * 64;
  const int n_base = blockIdx.y * 64;
  const int w  = tid >> 5;
  const int wm = (w >> 1) * 32;
  const int wn = (w & 1) * 32;

  // A staging: 512 float4 elements, 4 per thread (row-contiguous in k).
  const int arow = tid >> 3;          // 0..15 (+16 per i)
  const int acol = (tid & 7) * 4;
  // B staging: thread owns one LDS column, k-contiguous run of 16.
  //   n = tid & 63, kh = (tid>>6)*16. Global side: 16 lane-coalesced b32
  //   loads (W is L2-resident); LDS side: 2x b128 stores.
  const int bn = tid & 63;
  const int bkh = (tid >> 6) * 16;

  const float* pA = A + (size_t)(m_base + arow) * K + acol;
  const float* pW = W + (size_t)bkh * N + n_base + bn;

  float4 ra[4];
  float  rw[16];
#pragma unroll
  for (int i = 0; i < 4; ++i)          // prefetch kk = 0 (A)
    ra[i] = *(const float4*)(pA + (size_t)i * 16 * K);
#pragma unroll
  for (int j = 0; j < 16; ++j)         // prefetch kk = 0 (W, strided by N)
    rw[j] = pW[(size_t)j * N];

  v8f acc00 = {}, acc01 = {}, acc10 = {}, acc11 = {};

  for (int kk = 0; kk < K; kk += 32) {
    // Store staged registers to LDS (f32 -> f16 at store time).
#pragma unroll
    for (int i = 0; i < 4; ++i) {
      h4 av;
      av[0] = (h16)ra[i].x; av[1] = (h16)ra[i].y;
      av[2] = (h16)ra[i].z; av[3] = (h16)ra[i].w;
      *(h4*)(As + (arow + i * 16) * 48 + acol) = av;
    }
    {
      h8 b0, b1;
#pragma unroll
      for (int j = 0; j < 8; ++j) { b0[j] = (h16)rw[j]; b1[j] = (h16)rw[j + 8]; }
      h16* d = Bs + bn * 48 + bkh;
      *(h8*)(d) = b0;
      *(h8*)(d + 8) = b1;
    }
    __syncthreads();

    const int kk2 = kk + 32;
    if (kk2 < K) {                      // issue next K-step's loads now
#pragma unroll
      for (int i = 0; i < 4; ++i)
        ra[i] = *(const float4*)(pA + (size_t)i * 16 * K + kk2);
#pragma unroll
      for (int j = 0; j < 16; ++j)
        rw[j] = pW[(size_t)(kk2 + j) * N];
    }

    v16h a0 = load_frag_a(As + wm * 48, 48);
    v16h a1 = load_frag_a(As + (wm + 16) * 48, 48);
    v16h b0 = load_frag_b(Bs + wn * 48, 48, 0);
    v16h b1 = load_frag_b(Bs + (wn + 16) * 48, 48, 0);
    acc00 = wmma_f16(a0, b0, acc00);
    acc01 = wmma_f16(a0, b1, acc01);
    acc10 = wmma_f16(a1, b0, acc10);
    acc11 = wmma_f16(a1, b1, acc11);
    __syncthreads();
  }

  const int lane = tid & 31;
  const int ln = lane & 15;
  const int hi = lane >> 4;
  v8f accs[2][2] = {{acc00, acc01}, {acc10, acc11}};
#pragma unroll
  for (int ti = 0; ti < 2; ++ti)
#pragma unroll
    for (int tj = 0; tj < 2; ++tj) {
      v8f acc = accs[ti][tj];
#pragma unroll
      for (int r = 0; r < 8; ++r) {
        int m = m_base + wm + ti * 16 + r + hi * 8;
        int n = n_base + wn + tj * 16 + ln;
        float val = acc[r] + bias[n];
        if (mode == 0) {
          // Final output: streamed once, never re-read -> NT store.
          __builtin_nontemporal_store(val, outF + (size_t)m * N + n);
        } else {
          int b = m >> 11, s = m & (SEQ - 1);   // SEQ = 2048
          int h = n >> 6, d = n & (DK - 1);
          size_t idx = (mode == 1)
              ? (((size_t)(b * NUM_HEADS + h) * SEQ + s) * DK + d)
              : (((size_t)(b * NUM_HEADS + h) * DK + d) * SEQ + s);
          outH[idx] = (h16)val;                 // re-read by attention: keep RT
        }
      }
    }
}

// ---------------------------------------------------------------------------
// Attention: one block = one (b,h) head x 16 query rows. 128 threads (4 waves).
// Dynamic LDS: f32 scores 16xSEQ (128KB) + f16 probs 16xSEQ (64KB) + reduce.
// attn goes to HBM once (NT stores); the PV GEMM reads probabilities from LDS
// and V (f16, L2-resident) from global.
// ---------------------------------------------------------------------------
__global__ __launch_bounds__(128) void attn_kernel(
    const h16* __restrict__ Qh, const h16* __restrict__ Kh,
    const h16* __restrict__ Vt, float* __restrict__ attnOut,
    float* __restrict__ combined) {
  extern __shared__ unsigned char smem[];
  float* Ps  = (float*)smem;                                   // 16 x PS_LD f32
  h16*   Phh = (h16*)(smem + (size_t)16 * PS_LD * 4);          // 16 x PH_LD f16
  float* red = (float*)(smem + (size_t)16 * PS_LD * 4 + (size_t)16 * PH_LD * 2);

  const int blk  = blockIdx.x;
  const int qblk = blk & (SEQ / 16 - 1);   // 0..127
  const int bh   = blk >> 7;               // 0..31
  const int q0   = qblk * 16;

  const h16* Qbase = Qh + ((size_t)bh * SEQ + q0) * DK;
  const h16* Kbase = Kh + (size_t)bh * SEQ * DK;
  const h16* Vbase = Vt + (size_t)bh * DK * SEQ;

  const int wv   = threadIdx.x >> 5;
  const int lane = threadIdx.x & 31;
  const int ln   = lane & 15;
  const int hi   = lane >> 4;
  const float scale = 0.125f;   // 1/sqrt(64)

  // ---- Phase 1: scores = scale * Q K^T, into LDS ----
  v16h qf0 = load_frag_a(Qbase + 0, DK);
  v16h qf1 = load_frag_a(Qbase + 32, DK);
  for (int tt = wv; tt < SEQ / 16; tt += 4) {
    int col0 = tt * 16;
    v16h kf0 = load_frag_b(Kbase + (size_t)col0 * DK, DK, 0);
    v16h kf1 = load_frag_b(Kbase + (size_t)col0 * DK, DK, 32);
    v8f acc = {};
    acc = wmma_f16(qf0, kf0, acc);
    acc = wmma_f16(qf1, kf1, acc);
#pragma unroll
    for (int r = 0; r < 8; ++r)
      Ps[(r + hi * 8) * PS_LD + col0 + ln] = acc[r] * scale;
  }
  __syncthreads();

  // ---- Phase 2: fp32 softmax (thread = (row r, 256-col chunk p)) ----
  const int r  = threadIdx.x & 15;
  const int p  = threadIdx.x >> 4;         // 0..7
  const int c0 = p * (SEQ / 8);
  float* prow = Ps + r * PS_LD + c0;

  float mx = -1e30f;
  for (int j = 0; j < SEQ / 8; j += 4) {
    f4v v = *(const f4v*)(prow + j);
    mx = fmaxf(mx, fmaxf(fmaxf(v[0], v[1]), fmaxf(v[2], v[3])));
  }
  red[r * 8 + p] = mx;
  __syncthreads();
  float rmx = red[r * 8 + 0];
#pragma unroll
  for (int i = 1; i < 8; ++i) rmx = fmaxf(rmx, red[r * 8 + i]);
  __syncthreads();

  float sum = 0.f;
  for (int j = 0; j < SEQ / 8; j += 4) {
    f4v v = *(const f4v*)(prow + j);
    f4v e;
    e[0] = expf(v[0] - rmx); e[1] = expf(v[1] - rmx);
    e[2] = expf(v[2] - rmx); e[3] = expf(v[3] - rmx);
    *(f4v*)(prow + j) = e;
    sum += (e[0] + e[1]) + (e[2] + e[3]);
  }
  red[r * 8 + p] = sum;
  __syncthreads();
  float tot = 0.f;
#pragma unroll
  for (int i = 0; i < 8; ++i) tot += red[r * 8 + i];
  const float inv = 1.0f / tot;

  float* arow_p = attnOut + ((size_t)bh * SEQ + q0 + r) * SEQ + c0;
  h16*   hrow_p = Phh + r * PH_LD + c0;
  for (int j = 0; j < SEQ / 8; j += 4) {
    f4v v = *(const f4v*)(prow + j);
    v[0] *= inv; v[1] *= inv; v[2] *= inv; v[3] *= inv;
    __builtin_nontemporal_store(v, (f4v*)(arow_p + j));   // single HBM write
    h4 hv; hv[0] = (h16)v[0]; hv[1] = (h16)v[1];
    hv[2] = (h16)v[2]; hv[3] = (h16)v[3];
    *(h4*)(hrow_p + j) = hv;
  }
  __syncthreads();

  // ---- Phase 3: ctx = P @ V, wave wv owns d-tile d0 = wv*16.
  // Two interleaved accumulator chains break the WMMA RAW latency chain.
  const int d0 = wv * 16;
  v8f accA = {}, accB = {};
  for (int ks = 0; ks < SEQ; ks += 64) {
    v16h af0 = load_frag_a(Phh + ks, PH_LD);
    v16h bf0 = load_frag_b(Vbase + (size_t)d0 * SEQ + ks, SEQ, 0);
    accA = wmma_f16(af0, bf0, accA);
    v16h af1 = load_frag_a(Phh + ks + 32, PH_LD);
    v16h bf1 = load_frag_b(Vbase + (size_t)d0 * SEQ + ks + 32, SEQ, 0);
    accB = wmma_f16(af1, bf1, accB);
  }
  v8f acc = accA + accB;

  const int b = bh >> 4, h = bh & 15;
#pragma unroll
  for (int rr = 0; rr < 8; ++rr) {
    int m = rr + hi * 8;
    combined[((size_t)b * SEQ + q0 + m) * D_MODEL + h * DK + d0 + ln] = acc[rr];
  }
}

// ---------------------------------------------------------------------------
extern "C" void kernel_launch(void* const* d_in, const int* in_sizes, int n_in,
                              void* d_out, int out_size, void* d_ws,
                              size_t ws_size, hipStream_t stream) {
  const float* query = (const float*)d_in[0];
  const float* key_  = (const float*)d_in[1];
  const float* value = (const float*)d_in[2];
  const float* W_q   = (const float*)d_in[3];
  const float* b_q   = (const float*)d_in[4];
  const float* W_k   = (const float*)d_in[5];
  const float* b_k   = (const float*)d_in[6];
  const float* W_v   = (const float*)d_in[7];
  const float* b_v   = (const float*)d_in[8];
  const float* W_o   = (const float*)d_in[9];
  const float* b_o   = (const float*)d_in[10];

  float* out  = (float*)d_out;                      // [B*S, D] f32
  float* attn = out + (size_t)MTOT * D_MODEL;       // [B,H,S,S] f32

  // Workspace (40 MB): Qh/Kh f16 head-major, Vt f16 [bh][d][s], combined f32.
  h16* Qh = (h16*)d_ws;
  h16* Kh = Qh + (size_t)MTOT * D_MODEL;
  h16* Vt = Kh + (size_t)MTOT * D_MODEL;
  float* combined = (float*)(Vt + (size_t)MTOT * D_MODEL);

  dim3 gridP(MTOT / 64, D_MODEL / 64);
  dim3 blockP(128);
  gemm_proj<<<gridP, blockP, 0, stream>>>(query, W_q, b_q, nullptr, Qh,
                                          MTOT, D_MODEL, D_MODEL, 1);
  gemm_proj<<<gridP, blockP, 0, stream>>>(key_, W_k, b_k, nullptr, Kh,
                                          MTOT, D_MODEL, D_MODEL, 1);
  gemm_proj<<<gridP, blockP, 0, stream>>>(value, W_v, b_v, nullptr, Vt,
                                          MTOT, D_MODEL, D_MODEL, 2);

  size_t lds = (size_t)16 * PS_LD * 4 + (size_t)16 * PH_LD * 2 + 16 * 8 * 4;
  (void)hipFuncSetAttribute((const void*)attn_kernel,
                            hipFuncAttributeMaxDynamicSharedMemorySize,
                            (int)lds);
  attn_kernel<<<dim3(BATCH * NUM_HEADS * (SEQ / 16)), dim3(128), lds, stream>>>(
      Qh, Kh, Vt, attn, combined);

  gemm_proj<<<gridP, blockP, 0, stream>>>(combined, W_o, b_o, out, nullptr,
                                          MTOT, D_MODEL, D_MODEL, 0);
}
